// SOGNN_41918880809097
// MI455X (gfx1250) — compile-verified, matrix operand order checked by reading, not run on hardware
//
#include <hip/hip_runtime.h>
#include <hip/hip_bf16.h>
#include <math.h>

// ---------------------------------------------------------------------------
// SOGNN on MI455X (gfx1250): bf16 WMMA (v_wmma_f32_16x16x32_bf16) for all
// GEMM-shaped work, fp32 VALU for conv1 / tanh / softmax / top-k.
// ~50 GFLOP, ~100MB working set (fits in 192MB L2) -> matrix-core bound.
// Round-2 fixes: remove per-element guards in im2col gather (pad LDS instead;
// garbage B-columns only affect discarded D-columns) -> kills the
// s_and_saveexec branch storm; pad cls_w to 16 rows so the classifier B
// fragment needs no row guard; prefetch conv weights.
// Round-3: fix missing 'out' declaration in attn pad-row branch.
// ---------------------------------------------------------------------------

typedef __attribute__((ext_vector_type(16))) __bf16 v16bf;
typedef __attribute__((ext_vector_type(8)))  float  v8f;

#define BATCH 512
#define E_NODES 62
#define E_PAD 64
#define BE (BATCH * E_NODES)       // 31744
#define NODE_DIM 512

__device__ __forceinline__ v8f wmma_bf16(v16bf a, v16bf b, v8f c) {
  // D = A(16x32) * B(32x16) + C, f32 accumulate
  return __builtin_amdgcn_wmma_f32_16x16x32_bf16(
      /*neg_a=*/false, a, /*neg_b=*/false, b,
      /*c_mod=*/(short)0, c, /*reuse_a=*/false, /*reuse_b=*/false);
}

// --- Fragment builders (CDNA5 wave32 16-bit A/B layout) ---------------------
// A-matrix 16x32 (rows x K), row-major source with row stride ld:
//   lane<16 : row=lane,    K = {0..7} U {16..23}
//   lane>=16: row=lane-16, K = {8..15} U {24..31}
// Each half is 8 contiguous bf16 (16 bytes) -> coalesces to b128 loads.
// Same layout serves B when B is supplied transposed (N x K row-major).
__device__ __forceinline__ v16bf frag_rm(const __bf16* base, int ld) {
  const int lane = threadIdx.x & 31;
  const int r = lane & 15, hi = lane >> 4;
  const __bf16* p = base + (size_t)r * ld + hi * 8;
  v16bf f;
#pragma unroll
  for (int i = 0; i < 8; ++i) f[i] = p[i];
#pragma unroll
  for (int i = 0; i < 8; ++i) f[8 + i] = p[16 + i];
  return f;
}

// B fragment from a K-major (K x N row-major) tile: gather with stride ld.
__device__ __forceinline__ v16bf frag_km(const __bf16* base, int ld) {
  const int lane = threadIdx.x & 31;
  const int n = lane & 15, hi = lane >> 4;
  v16bf f;
#pragma unroll
  for (int i = 0; i < 8; ++i) f[i] = base[(size_t)(hi * 8 + i) * ld + n];
#pragma unroll
  for (int i = 0; i < 8; ++i) f[8 + i] = base[(size_t)(16 + hi * 8 + i) * ld + n];
  return f;
}

// im2col B fragment for 1-D conv (kernel width 5) from LDS activations [C][W].
// kt MUST be compile-time constant at every call site (callers fully unroll)
// so c = k/5, kw = k%5 fold -> unconditional ds_load_u16 with const offsets.
// NO bounds guard: out-of-range n reads padded LDS garbage, which corrupts
// only D-columns that the epilogue discards (D[:,n] depends only on B[:,n]).
__device__ __forceinline__ v16bf conv_bfrag(const __bf16* src, int W, int kt, int nbase) {
  const int lane = threadIdx.x & 31;
  const int n = (lane & 15) + nbase;
  const int hi = lane >> 4;
  const __bf16* rowp = src + n;          // per-lane base; offsets are constants
  v16bf f;
#pragma unroll
  for (int i = 0; i < 16; ++i) {
    const int kk = (i < 8) ? (hi * 8 + i) : (16 + hi * 8 + (i - 8));
    const int k = kt * 32 + kk;
    f[i] = rowp[(k / 5) * W + (k % 5)];
  }
  return f;
}

// C/D (16x16 f32) layout: n = lane&15, m = r + 8*(lane>>4)
__device__ __forceinline__ void store_tile_f32(float* base, int ld, v8f c, int ncols) {
  const int lane = threadIdx.x & 31;
  const int n = lane & 15, half = lane >> 4;
  if (n < ncols) {
#pragma unroll
    for (int r = 0; r < 8; ++r) base[(size_t)(r + 8 * half) * ld + n] = c[r];
  }
}
__device__ __forceinline__ void store_tile_bf16(__bf16* base, int ld, v8f c, int ncols) {
  const int lane = threadIdx.x & 31;
  const int n = lane & 15, half = lane >> 4;
  if (n < ncols) {
#pragma unroll
    for (int r = 0; r < 8; ++r) base[(size_t)(r + 8 * half) * ld + n] = (__bf16)c[r];
  }
}

// ---------------------------------------------------------------------------
// Weight f32 -> bf16 conversion / zero fill
// ---------------------------------------------------------------------------
__global__ void cvt_kernel(const float* __restrict__ src, __bf16* __restrict__ dst, int n) {
  int i = blockIdx.x * 256 + threadIdx.x;
  if (i < n) dst[i] = (__bf16)src[i];
}
__global__ void zfill_kernel(__bf16* __restrict__ dst, int n) {
  int i = blockIdx.x * 256 + threadIdx.x;
  if (i < n) dst[i] = (__bf16)0.f;
}

// Zero padded node rows (e = 62,63) of H so padded GEMM tiles are exact.
__global__ void pad_kernel(__bf16* __restrict__ Hbf) {
  int i = blockIdx.x * 256 + threadIdx.x;   // over BATCH*2*512
  if (i < BATCH * 2 * NODE_DIM) {
    int b = i >> 10;
    int r = (i >> 9) & 1;
    int d = i & (NODE_DIM - 1);
    Hbf[((size_t)(b * E_PAD + E_NODES + r)) * NODE_DIM + d] = (__bf16)0.f;
  }
}

// ---------------------------------------------------------------------------
// Fused conv chain: conv1 (VALU) -> conv2 (WMMA) -> conv3 (WMMA) per image.
// 4 waves/block, one 5x64 image per wave. ~46KB LDS per block.
// Input staged with GLOBAL_LOAD_ASYNC_TO_LDS (ASYNCcnt path, CDNA5 §08).
// c1/c2 are padded so unguarded im2col gathers stay in-bounds.
// ---------------------------------------------------------------------------
struct __align__(16) ConvSmem {
  float  in[4][5 * 64];        // staged input (async-to-LDS)
  __bf16 c1[4][32 * 30 + 64];  // conv1 out (relu+pool), + gather padding
  float  craw[4][64 * 26];     // pre-pool scratch (reused by conv2/conv3)
  __bf16 c2[4][64 * 13 + 32];  // conv2 out (relu+pool), + gather padding
};

__global__ __launch_bounds__(128) void conv_chain_kernel(
    const float* __restrict__ x,
    const float* __restrict__ w1, const float* __restrict__ b1,
    const __bf16* __restrict__ w2, const float* __restrict__ b2,
    const __bf16* __restrict__ w3, const float* __restrict__ b3,
    __bf16* __restrict__ Hbf) {
  __shared__ ConvSmem sm;
  const int wave = threadIdx.x >> 5;
  const int lane = threadIdx.x & 31;
  const int img = blockIdx.x * 4 + wave;      // BE = 7936*4 exactly

  // warm L2/L0 with conv weights while conv1 runs (global_prefetch_b8)
  for (int i = threadIdx.x; i < 400; i += 128)     // 400 x 256B lines (w3 itself is 128*320*2B = 320 lines)
    __builtin_prefetch((const char*)w3 + i * 256, 0, 1);
  for (int i = threadIdx.x; i < 80; i += 128)      // 64*160*2B / 256B lines
    __builtin_prefetch((const char*)w2 + i * 256, 0, 1);

  // stage input 5x64 f32 = 80 x 16B, asynchronously into LDS
  {
    const float* xin = x + (size_t)img * 320;
#pragma unroll
    for (int i0 = 0; i0 < 80; i0 += 32) {
      const int i = i0 + lane;
      if (i < 80) {
        unsigned lds_addr = (unsigned)(unsigned long long)(&sm.in[wave][i * 4]);
        unsigned long long gaddr = (unsigned long long)(xin + i * 4);
        asm volatile("global_load_async_to_lds_b128 %0, %1, off"
                     :: "v"(lds_addr), "v"(gaddr) : "memory");
      }
    }
    asm volatile("s_wait_asynccnt 0x0" ::: "memory");
  }
  __syncthreads();

  // conv1: K=25 too small for WMMA; VALU. out 32x60 -> relu -> pool -> 32x30
  // lane = pooled output column wp (lanes 30,31 idle) -> no int division.
  if (lane < 30) {
    const int wp = lane;
#pragma unroll 4
    for (int o = 0; o < 32; ++o) {
      float mx = -3.4e38f;
#pragma unroll
      for (int p = 0; p < 2; ++p) {
        const int w0 = 2 * wp + p;
        float s = b1[o];
#pragma unroll
        for (int kh = 0; kh < 5; ++kh)
#pragma unroll
          for (int kw = 0; kw < 5; ++kw)
            s += sm.in[wave][kh * 64 + w0 + kw] * w1[o * 25 + kh * 5 + kw];
        mx = fmaxf(mx, s);
      }
      sm.c1[wave][o * 30 + wp] = (__bf16)fmaxf(mx, 0.f);
    }
  }
  __syncthreads();

  // conv2 GEMM: M=64 (4 mtiles), N=26 (2 ntiles), K=160 (5 ktiles, unrolled)
  {
    v8f acc[4][2] = {};
#pragma unroll
    for (int kt = 0; kt < 5; ++kt) {
      v16bf bf0 = conv_bfrag(sm.c1[wave], 30, kt, 0);
      v16bf bf1 = conv_bfrag(sm.c1[wave], 30, kt, 16);
#pragma unroll
      for (int mt = 0; mt < 4; ++mt) {
        v16bf a = frag_rm(w2 + (size_t)mt * 16 * 160 + kt * 32, 160);
        acc[mt][0] = wmma_bf16(a, bf0, acc[mt][0]);
        acc[mt][1] = wmma_bf16(a, bf1, acc[mt][1]);
      }
    }
    const int n = lane & 15, half = lane >> 4;
#pragma unroll
    for (int mt = 0; mt < 4; ++mt)
#pragma unroll
      for (int nt = 0; nt < 2; ++nt) {
        const int nn = nt * 16 + n;
        if (nn < 26) {
#pragma unroll
          for (int r = 0; r < 8; ++r) {
            const int o = mt * 16 + r + 8 * half;
            sm.craw[wave][o * 26 + nn] = fmaxf(acc[mt][nt][r] + b2[o], 0.f);
          }
        }
      }
  }
  __syncthreads();
  // pool 26 -> 13: lane = pooled column (lanes 13..31 idle), no division
  if (lane < 13) {
#pragma unroll 8
    for (int o = 0; o < 64; ++o)
      sm.c2[wave][o * 13 + lane] =
          (__bf16)fmaxf(sm.craw[wave][o * 26 + 2 * lane], sm.craw[wave][o * 26 + 2 * lane + 1]);
  }
  __syncthreads();

  // conv3 GEMM: M=128 (8 mtiles), N=9 (1 ntile), K=320 (10 ktiles, unrolled)
  {
    v8f acc[8] = {};
#pragma unroll
    for (int kt = 0; kt < 10; ++kt) {
      v16bf bfr = conv_bfrag(sm.c2[wave], 13, kt, 0);
#pragma unroll
      for (int mt = 0; mt < 8; ++mt) {
        v16bf a = frag_rm(w3 + (size_t)mt * 16 * 320 + kt * 32, 320);
        acc[mt] = wmma_bf16(a, bfr, acc[mt]);
      }
    }
    const int n = lane & 15, half = lane >> 4;
    if (n < 9) {
#pragma unroll
      for (int mt = 0; mt < 8; ++mt)
#pragma unroll
        for (int r = 0; r < 8; ++r) {
          const int c = mt * 16 + r + 8 * half;
          sm.craw[wave][c * 9 + n] = fmaxf(acc[mt][r] + b3[c], 0.f);
        }
    }
  }
  __syncthreads();

  // pool 9 -> 4, write H row (d = c*4 + wp), padded node layout (B,64,512)
  const int b = img / E_NODES, e = img % E_NODES;
  __bf16* hrow = Hbf + ((size_t)b * E_PAD + e) * NODE_DIM;
#pragma unroll 4
  for (int idx = lane; idx < NODE_DIM; idx += 32) {
    const int c = idx >> 2, wp = idx & 3;
    hrow[idx] = (__bf16)fmaxf(sm.craw[wave][c * 9 + 2 * wp], sm.craw[wave][c * 9 + 2 * wp + 1]);
  }
}

// ---------------------------------------------------------------------------
// G = tanh(H @ Wbn^T) : rows = BATCH*64 (padded), N=32, K=Din
// ---------------------------------------------------------------------------
__global__ __launch_bounds__(256) void gemm_tanh_kernel(
    const __bf16* __restrict__ Hin, int Din,
    const __bf16* __restrict__ Wbn,      // (32, Din) row-major == B^T
    __bf16* __restrict__ G) {            // (BATCH*64, 32)
  const int wave = threadIdx.x >> 5;
  const int strip = blockIdx.x * 8 + wave;     // 2048 strips of 16 rows
  const __bf16* A0 = Hin + (size_t)strip * 16 * Din;
  v8f acc[2] = {};
  const int ktiles = Din >> 5;
  for (int kt = 0; kt < ktiles; ++kt) {
    v16bf a = frag_rm(A0 + kt * 32, Din);
#pragma unroll
    for (int nt = 0; nt < 2; ++nt) {
      v16bf b = frag_rm(Wbn + (size_t)nt * 16 * Din + kt * 32, Din);
      acc[nt] = wmma_bf16(a, b, acc[nt]);
    }
  }
  const int lane = threadIdx.x & 31;
  const int n = lane & 15, half = lane >> 4;
#pragma unroll
  for (int nt = 0; nt < 2; ++nt)
#pragma unroll
    for (int r = 0; r < 8; ++r) {
      const int m = r + 8 * half;
      G[((size_t)strip * 16 + m) * 32 + nt * 16 + n] = (__bf16)tanhf(acc[nt][r]);
    }
}

// ---------------------------------------------------------------------------
// Per batch: S = G G^T (WMMA), row softmax over 62, top-10 sparsify -> A_sp
// ---------------------------------------------------------------------------
__global__ __launch_bounds__(256) void attn_kernel(
    const __bf16* __restrict__ G,     // (B,64,32)
    __bf16* __restrict__ Asp) {       // (B,64,64)
  __shared__ float S[E_PAD * E_PAD];
  const int b = blockIdx.x;
  const int wave = threadIdx.x >> 5;
  const __bf16* Gb = G + (size_t)b * E_PAD * 32;
#pragma unroll
  for (int t0 = 0; t0 < 16; t0 += 8) {     // 4x4 tiles, K=32 (one ktile)
    const int t = t0 + wave;
    const int mt = t >> 2, nt = t & 3;
    v16bf a = frag_rm(Gb + (size_t)mt * 16 * 32, 32);
    v16bf bb = frag_rm(Gb + (size_t)nt * 16 * 32, 32);
    v8f c = {};
    c = wmma_bf16(a, bb, c);
    store_tile_f32(&S[(mt * 16) * E_PAD + nt * 16], E_PAD, c, 16);
  }
  __syncthreads();
  const int t = threadIdx.x;
  if (t < E_NODES) {
    float row[E_NODES];
    float mx = -3.4e38f;
#pragma unroll 1
    for (int f = 0; f < E_NODES; ++f) { row[f] = S[t * E_PAD + f]; mx = fmaxf(mx, row[f]); }
    float sum = 0.f;
#pragma unroll 1
    for (int f = 0; f < E_NODES; ++f) { row[f] = __expf(row[f] - mx); sum += row[f]; }
    const float inv = 1.f / sum;
    unsigned long long picked = 0ull;
#pragma unroll 1
    for (int j = 0; j < 10; ++j) {          // top-10 (softmax is monotone in exp)
      int bi = 0; float bv = -1.f;
#pragma unroll 1
      for (int f = 0; f < E_NODES; ++f)
        if (!((picked >> f) & 1ull) && row[f] > bv) { bv = row[f]; bi = f; }
      picked |= 1ull << bi;
    }
    __bf16* out = Asp + ((size_t)b * E_PAD + t) * E_PAD;
#pragma unroll 1
    for (int f = 0; f < E_PAD; ++f) {
      const float v = (f < E_NODES && ((picked >> f) & 1ull)) ? row[f] * inv : 0.f;
      out[f] = (__bf16)v;
    }
  } else if (t < E_PAD) {
    __bf16* out = Asp + ((size_t)b * E_PAD + t) * E_PAD;
#pragma unroll 1
    for (int f = 0; f < E_PAD; ++f) out[f] = (__bf16)0.f;
  }
}

// ---------------------------------------------------------------------------
// Fused: Mstrip(16 x Din) = A_sp[strip rows] @ H ; Hout = relu(M @ gc_w^T + b)
// grid = BATCH * 4 strips; M kept in LDS (never materialized globally).
// ---------------------------------------------------------------------------
__global__ __launch_bounds__(256) void sogc_kernel(
    const __bf16* __restrict__ Asp,   // (B,64,64)
    const __bf16* __restrict__ Hin,   // (B,64,Din)
    int Din,
    const __bf16* __restrict__ Gw,    // (64, Din) == B^T
    const float* __restrict__ Gbias,  // (64)
    __bf16* __restrict__ Hout) {      // (B,64,64)
  extern __shared__ __bf16 Ms[];      // 16 x Din
  const int b = blockIdx.x >> 2;
  const int strip = blockIdx.x & 3;
  const int wave = threadIdx.x >> 5;
  const __bf16* Arow = Asp + (size_t)b * E_PAD * E_PAD + (size_t)strip * 16 * E_PAD;
  const __bf16* Hb = Hin + (size_t)b * E_PAD * Din;

  // phase 1: M = A_sp @ H ; K = 64 (2 ktiles), B side gathered K-major from H
  const int ntiles = Din >> 4;
  for (int nt = wave; nt < ntiles; nt += 8) {
    v8f acc = {};
#pragma unroll
    for (int kt = 0; kt < 2; ++kt) {
      v16bf a = frag_rm(Arow + kt * 32, E_PAD);
      v16bf bb = frag_km(Hb + (size_t)kt * 32 * Din + nt * 16, Din);
      acc = wmma_bf16(a, bb, acc);
    }
    store_tile_bf16(&Ms[nt * 16], Din, acc, 16);
  }
  __syncthreads();

  // phase 2: Hout strip = relu(M @ gc_w^T + b); 4 ntiles -> waves 0..3
  if (wave < 4) {
    const int nt = wave;
    v8f acc = {};
    const int ktiles = Din >> 5;
    for (int kt = 0; kt < ktiles; ++kt) {
      v16bf a = frag_rm(&Ms[kt * 32], Din);
      v16bf bb = frag_rm(Gw + (size_t)nt * 16 * Din + kt * 32, Din);
      acc = wmma_bf16(a, bb, acc);
    }
    const int lane = threadIdx.x & 31;
    const int n = lane & 15, half = lane >> 4;
#pragma unroll
    for (int r = 0; r < 8; ++r) {
      const int e = strip * 16 + r + 8 * half;
      float v = fmaxf(acc[r] + Gbias[nt * 16 + n], 0.f);
      if (e >= E_NODES) v = 0.f;            // keep pad rows exactly zero
      Hout[((size_t)b * E_PAD + e) * E_PAD + nt * 16 + n] = (__bf16)v;
    }
  }
}

// ---------------------------------------------------------------------------
// logits = Hfin.reshape(B, 62*64) @ cls_w^T + b ; one wave per 16 batch rows.
// Padded row stride 4096; cls_w padded to 16 rows (rows 4..15 zero-filled),
// so the B fragment is a plain unguarded contiguous load.
// ---------------------------------------------------------------------------
__global__ __launch_bounds__(32) void cls_kernel(
    const __bf16* __restrict__ Hfin,  // (B,64,64)
    const __bf16* __restrict__ Wc,    // (16, 3968), rows >=4 are zero
    const float* __restrict__ bc,     // (4)
    float* __restrict__ out) {        // (512, 4)
  const int strip = blockIdx.x;
  const __bf16* A0 = Hfin + (size_t)strip * 16 * 4096;
  v8f acc = {};
  for (int kt = 0; kt < 124; ++kt) {
    v16bf a = frag_rm(A0 + kt * 32, 4096);
    v16bf bb = frag_rm(Wc + kt * 32, 3968);
    acc = wmma_bf16(a, bb, acc);
  }
  const int lane = threadIdx.x & 31;
  const int n = lane & 15, half = lane >> 4;
  if (n < 4) {
#pragma unroll
    for (int r = 0; r < 8; ++r) {
      const int m = r + 8 * half;
      out[(strip * 16 + m) * 4 + n] = acc[r] + bc[n];
    }
  }
}

// ---------------------------------------------------------------------------
extern "C" void kernel_launch(void* const* d_in, const int* in_sizes, int n_in,
                              void* d_out, int out_size, void* d_ws, size_t ws_size,
                              hipStream_t stream) {
  (void)in_sizes; (void)n_in; (void)out_size; (void)ws_size;
  const float* x  = (const float*)d_in[0];
  const float* w1 = (const float*)d_in[1];
  const float* b1 = (const float*)d_in[2];
  const float* b2 = (const float*)d_in[4];
  const float* b3 = (const float*)d_in[6];

  __bf16* ws = (__bf16*)d_ws;
  size_t off = 0;
  auto alloc = [&](size_t n) { __bf16* p = ws + off; off += (n + 7) & ~(size_t)7; return p; };
  __bf16* W2  = alloc(64 * 160);
  __bf16* W3  = alloc(128 * 320);
  __bf16* WB1 = alloc(32 * 512);
  __bf16* GC1 = alloc(64 * 512);
  __bf16* WB2 = alloc(32 * 64);
  __bf16* GC2 = alloc(64 * 64);
  __bf16* WB3 = alloc(32 * 64);
  __bf16* GC3 = alloc(64 * 64);
  __bf16* WCL = alloc(16 * 3968);                          // padded to 16 rows
  __bf16* Hbf = alloc((size_t)BATCH * E_PAD * NODE_DIM);   // 16.8M elems
  __bf16* G   = alloc((size_t)BATCH * E_PAD * 32);
  __bf16* Asp = alloc((size_t)BATCH * E_PAD * E_PAD);
  __bf16* H1  = alloc((size_t)BATCH * E_PAD * E_PAD);
  __bf16* H2  = alloc((size_t)BATCH * E_PAD * E_PAD);

  auto cvt = [&](const void* s, __bf16* d, int n) {
    cvt_kernel<<<(n + 255) / 256, 256, 0, stream>>>((const float*)s, d, n);
  };
  cvt(d_in[3],  W2,  64 * 160);
  cvt(d_in[5],  W3,  128 * 320);
  cvt(d_in[7],  WB1, 32 * 512);
  cvt(d_in[8],  GC1, 64 * 512);
  cvt(d_in[10], WB2, 32 * 64);
  cvt(d_in[11], GC2, 64 * 64);
  cvt(d_in[13], WB3, 32 * 64);
  cvt(d_in[14], GC3, 64 * 64);
  cvt(d_in[16], WCL, 4 * 3968);
  zfill_kernel<<<(12 * 3968 + 255) / 256, 256, 0, stream>>>(WCL + 4 * 3968, 12 * 3968);

  pad_kernel<<<(BATCH * 2 * NODE_DIM + 255) / 256, 256, 0, stream>>>(Hbf);

  conv_chain_kernel<<<BE / 4, 128, 0, stream>>>(x, w1, b1, W2, b2, W3, b3, Hbf);

  // SOGC layer 1 (Din = 512)
  gemm_tanh_kernel<<<256, 256, 0, stream>>>(Hbf, 512, WB1, G);
  attn_kernel<<<BATCH, 256, 0, stream>>>(G, Asp);
  sogc_kernel<<<BATCH * 4, 256, 16 * 512 * sizeof(__bf16), stream>>>(
      Asp, Hbf, 512, GC1, (const float*)d_in[9], H1);

  // SOGC layer 2 (Din = 64)
  gemm_tanh_kernel<<<256, 256, 0, stream>>>(H1, 64, WB2, G);
  attn_kernel<<<BATCH, 256, 0, stream>>>(G, Asp);
  sogc_kernel<<<BATCH * 4, 256, 16 * 64 * sizeof(__bf16), stream>>>(
      Asp, H1, 64, GC2, (const float*)d_in[12], H2);

  // SOGC layer 3 (Din = 64)
  gemm_tanh_kernel<<<256, 256, 0, stream>>>(H2, 64, WB3, G);
  attn_kernel<<<BATCH, 256, 0, stream>>>(G, Asp);
  sogc_kernel<<<BATCH * 4, 256, 16 * 64 * sizeof(__bf16), stream>>>(
      Asp, H2, 64, GC3, (const float*)d_in[15], H1);

  cls_kernel<<<BATCH / 16, 32, 0, stream>>>(H1, WCL, (const float*)d_in[17], (float*)d_out);
}
